// ManifoldGNNPhysicsODE_1013612281973
// MI455X (gfx1250) — matrix operand order repaired and checked by reading, not compiled, not run on hardware
//
#include <hip/hip_runtime.h>

#define B_TOTAL 262144
#define NPERS   10000
#define PFEAT   16
#define DIM     64
#define NZONE   8

typedef __attribute__((ext_vector_type(16))) _Float16 v16h;
typedef __attribute__((ext_vector_type(8)))  _Float16 v8h;
typedef __attribute__((ext_vector_type(4)))  _Float16 v4h;
typedef __attribute__((ext_vector_type(8)))  float    v8f;
typedef __attribute__((ext_vector_type(4)))  unsigned int u32x4;
typedef __attribute__((ext_vector_type(8)))  int          i32x8;
typedef __attribute__((ext_vector_type(4)))  int          i32x4;

#if defined(__has_builtin)
#if __has_builtin(__builtin_amdgcn_tensor_load_to_lds) && __has_builtin(__builtin_amdgcn_s_wait_tensorcnt)
#define HAVE_TDM 1
#endif
#endif

// ---------------- LDS layout (bytes, all 16B aligned) ----------------
// Fragment region MUST stay at offset 0: the TDM descriptor targets lds_addr=0.
constexpr int LDS_WF1  = 0;                  // 20480 halfs: fused [160x128] frags (8nt x 5ks)
constexpr int LDS_WF2  = LDS_WF1 + 40960;    // 8192 halfs: [128x64] frags (4nt x 4ks)
constexpr int LDS_WF3  = LDS_WF2 + 16384;    // 1024 halfs: [64x8->16] frags (1nt x 2ks)
constexpr int LDS_ZET  = LDS_WF3 + 2048;     // 1024 halfs: ze^T [64x8->16] frags (1nt x 2ks)
constexpr int LDS_WT1  = LDS_ZET + 2048;     // 32 f32
constexpr int LDS_BT1  = LDS_WT1 + 128;      // 32 f32
constexpr int LDS_BF1  = LDS_BT1 + 128;      // 128 f32 (fused bf1')
constexpr int LDS_BF2  = LDS_BF1 + 512;      // 64 f32
constexpr int LDS_BF3  = LDS_BF2 + 256;      // 16 f32 (padded)
constexpr int LDS_ZE   = LDS_BF3 + 64;       // 8x64 f32
constexpr int LDS_ZN   = LDS_ZE + 2048;      // 8 f32 ||ze_z||^2 (pad to 64B)
constexpr int LDS_ADJ  = LDS_ZN + 64;        // 8x8 f32
constexpr int LDS_COMB = LDS_ADJ + 256;      // 8 waves x 16x160 halfs
constexpr int LDS_H1   = LDS_COMB + 40960;   // 8 waves x 16x128 halfs
constexpr int LDS_H2   = LDS_H1 + 32768;     // 8 waves x 16x64 halfs
constexpr int LDS_DES  = LDS_H2 + 16384;     // 8 waves x 16x16 f32 (desired flows)
constexpr int LDS_GW   = LDS_DES + 8192;     // 8 waves x 16x16 f32 (state . ze^T)
constexpr int LDS_TOTAL= LDS_GW + 8192;      // ~167.4 KB (<320 KB/WGP)

constexpr int FRAG_BYTES = 61440;            // 30720 halfs staged by TDM

// ---- workspace layout ----
// peF16:  640000 halfs @ 0
// frags:  30720 halfs  @ byte 1280000
// W1cat:  20480 f32    @ byte 1341440
// bf1p:   128 f32      @ byte 1423360

// ------------- prep 1: person embedding MLP -> f16 table -------------
__global__ void person_embed_kernel(const float* __restrict__ px,
                                    const float* __restrict__ Wp1,
                                    const float* __restrict__ bp1,
                                    const float* __restrict__ Wp2,
                                    const float* __restrict__ bp2,
                                    _Float16* __restrict__ peOut) {
  __shared__ float x[PFEAT];
  __shared__ float hid[64];
  const int p = blockIdx.x;
  const int tid = threadIdx.x;   // 64 threads
  if (tid < PFEAT) x[tid] = px[p * PFEAT + tid];
  __syncthreads();
  float acc = bp1[tid];
  #pragma unroll
  for (int i = 0; i < PFEAT; ++i) acc = fmaf(x[i], Wp1[i * 64 + tid], acc);
  hid[tid] = fmaxf(acc, 0.0f);
  __syncthreads();
  float o = bp2[tid];
  #pragma unroll
  for (int h = 0; h < 64; ++h) o = fmaf(hid[h], Wp2[h * 64 + tid], o);
  peOut[p * 64 + tid] = (_Float16)o;
}

// ------------- prep 2: fold time-encoder second layer into Wf1 -------
__global__ void fuse_time_kernel(const float* __restrict__ Wf1,
                                 const float* __restrict__ bf1,
                                 const float* __restrict__ Wt2,
                                 const float* __restrict__ bt2,
                                 float* __restrict__ W1cat,
                                 float* __restrict__ bf1p) {
  int gid = blockIdx.x * blockDim.x + threadIdx.x;
  if (gid < 16384) {
    W1cat[gid] = Wf1[gid];
  } else if (gid < 20480) {
    int e = gid - 16384;
    int j = e >> 7, n = e & 127;
    float acc = 0.0f;
    #pragma unroll 8
    for (int d = 0; d < 64; ++d) acc = fmaf(Wt2[j * 64 + d], Wf1[(128 + d) * 128 + n], acc);
    W1cat[(128 + j) * 128 + n] = acc;
  } else if (gid < 20608) {
    int n = gid - 20480;
    float acc = bf1[n];
    #pragma unroll 8
    for (int d = 0; d < 64; ++d) acc = fmaf(bt2[d], Wf1[(128 + d) * 128 + n], acc);
    bf1p[n] = acc;
  }
}

// ------------- prep 3: swizzle into WMMA B-fragment layout -----------
__global__ void weight_swizzle_kernel(const float* __restrict__ W1cat,
                                      const float* __restrict__ Wf2,
                                      const float* __restrict__ Wf3,
                                      const float* __restrict__ zeG,
                                      _Float16* __restrict__ frags) {
  int gid = blockIdx.x * blockDim.x + threadIdx.x;
  if (gid >= 30720) return;
  const float* W; int Nlog, KS, lid, base, transposed = 0;
  if (gid < 20480)      { W = W1cat; Nlog = 128; KS = 5; lid = gid;         base = 0;     }
  else if (gid < 28672) { W = Wf2;   Nlog = 64;  KS = 4; lid = gid - 20480; base = 20480; }
  else if (gid < 29696) { W = Wf3;   Nlog = 8;   KS = 2; lid = gid - 28672; base = 28672; }
  else                  { W = zeG;   Nlog = 8;   KS = 2; lid = gid - 29696; base = 29696; transposed = 1; }
  int frag = lid >> 9, rem = lid & 511;
  int lane = rem >> 4, i = rem & 15;
  int nt = frag / KS, ks = frag - nt * KS;
  int kl = ((lane >> 4) << 3) + (i & 7) + ((i >> 3) << 4);   // table order: 8h+i (i<8), 16+8h+(i-8) (i>=8)
  int n  = nt * 16 + (lane & 15);
  int k  = ks * 32 + kl;
  float v = 0.0f;
  if (n < Nlog) v = transposed ? W[n * 64 + k] : W[k * Nlog + n];
  frags[base + lid] = (_Float16)v;
}

// ---------------- fragment loaders ----------------
__device__ __forceinline__ v16h load_afrag(const _Float16* base, int stride,
                                           int lane, int k0) {
  const int ar = lane & 15;
  const int ko = (lane >> 4) << 3;
  const _Float16* p = base + ar * stride + k0 + ko;
  v8h lo = *(const v8h*)(p);
  v8h hi = *(const v8h*)(p + 16);
  return __builtin_shufflevector(lo, hi, 0,1,2,3,4,5,6,7,8,9,10,11,12,13,14,15);
}

__device__ __forceinline__ v16h load_bfrag(const _Float16* fragBase, int fragIdx,
                                           int lane) {
  return *(const v16h*)(fragBase + (fragIdx << 9) + (lane << 4));
}

// ---------------- main fused kernel ----------------
__global__ __launch_bounds__(256) void ode_main_kernel(
    const float* __restrict__ state, const float* __restrict__ tvec,
    const int* __restrict__ person_idx,
    const float* __restrict__ zeG, const float* __restrict__ adjG,
    const float* __restrict__ Wt1g, const float* __restrict__ bt1g,
    const float* __restrict__ bf1pG, const float* __restrict__ bf2g,
    const float* __restrict__ bf3g,
    const _Float16* __restrict__ peF16, const _Float16* __restrict__ fragsG,
    float* __restrict__ out) {
  extern __shared__ char smem[];
  const int tid = threadIdx.x;

  // ---- stage 61440B of B-fragments into LDS via the Tensor Data Mover ----
#if HAVE_TDM
  if (tid < 32) {   // wave 0 issues one TDM DMA; EXEC-independent, TENSORcnt-tracked
    unsigned long long ga = (unsigned long long)fragsG;
    // D# group0: count=1 | lds_addr=0 | global_addr | type=2  (ISA 8.3)
    u32x4 g0 = { 1u, 0u,
                 (unsigned)(ga & 0xFFFFFFFFu),
                 (unsigned)((ga >> 32) & 0x01FFFFFFu) | 0x80000000u };
    // D# group1: data_size=8B; tensor_dim0=tile_dim0=7680; dim1=1 (ISA 8.4)
    i32x8 g1 = { 0x00030000,      // [17:16] data_size=3 (8B)
                 0x1E000000,      // [63:48] tensor_dim0.lo = 7680
                 0x00010000,      // [95:80] tensor_dim1.lo = 1
                 0x1E000000,      // [127:112] tile_dim0 = 7680
                 0x00000001,      // [143:128] tile_dim1 = 1
                 7680,            // [191:160] tensor_dim0_stride.lo
                 0, 0 };
    i32x4 g2 = { 0, 0, 0, 0 };
    i32x4 g3 = { 0, 0, 0, 0 };
    i32x8 g4 = { 0, 0, 0, 0, 0, 0, 0, 0 };
    __builtin_amdgcn_tensor_load_to_lds(g0, g1, g2, g3, g4, 0);
    __builtin_amdgcn_s_wait_tensorcnt(0);
  }
#else
  {
    uint4* dstF = (uint4*)(smem + LDS_WF1);
    const uint4* srcF = (const uint4*)fragsG;
    for (int i = tid; i < FRAG_BYTES / 16; i += 256) dstF[i] = srcF[i];
  }
#endif

  // ---- stage small params (normal path) ----
  {
    float* f;
    f = (float*)(smem + LDS_WT1); for (int i = tid; i < 32;  i += 256) f[i] = Wt1g[i];
    f = (float*)(smem + LDS_BT1); for (int i = tid; i < 32;  i += 256) f[i] = bt1g[i];
    f = (float*)(smem + LDS_BF1); for (int i = tid; i < 128; i += 256) f[i] = bf1pG[i];
    f = (float*)(smem + LDS_BF2); for (int i = tid; i < 64;  i += 256) f[i] = bf2g[i];
    f = (float*)(smem + LDS_ZE);  for (int i = tid; i < 512; i += 256) f[i] = zeG[i];
    f = (float*)(smem + LDS_ADJ); for (int i = tid; i < 64;  i += 256) f[i] = adjG[i];
    f = (float*)(smem + LDS_BF3); if (tid < 16) f[tid] = (tid < 8) ? bf3g[tid] : 0.0f;
    f = (float*)(smem + LDS_ZN);
    if (tid < 8) {
      float s = 0.0f;
      #pragma unroll 8
      for (int d = 0; d < 64; ++d) { float v = zeG[tid * 64 + d]; s = fmaf(v, v, s); }
      f[tid] = s;
    }
  }
  __syncthreads();

  const int wave = tid >> 5;
  const int lane = tid & 31;
  const int rowBase = blockIdx.x * 128 + wave * 16;
  const int r = lane & 15;
  const int h = lane >> 4;
  const int rowg = rowBase + r;
  const int ncol = lane & 15;
  const int m0 = (lane >> 4) << 3;

  _Float16* combW = (_Float16*)(smem + LDS_COMB) + wave * (16 * 160);
  _Float16* h1W   = (_Float16*)(smem + LDS_H1)   + wave * (16 * 128);
  _Float16* h2W   = (_Float16*)(smem + LDS_H2)   + wave * (16 * 64);
  float*    desW  = (float*)(smem + LDS_DES)     + wave * (16 * 16);
  float*    gW    = (float*)(smem + LDS_GW)      + wave * (16 * 16);

  // ---- build combined = [state | person_emb | relu(t*Wt1+bt1)] (16x160 f16) ----
  {
    _Float16* crow = combW + r * 160;
    const float4* sp = (const float4*)(state + (size_t)rowg * 64 + h * 32);
    #pragma unroll
    for (int q = 0; q < 8; ++q) {
      float4 v = sp[q];
      v4h hv = { (_Float16)v.x, (_Float16)v.y, (_Float16)v.z, (_Float16)v.w };
      *(v4h*)(crow + h * 32 + 4 * q) = hv;
    }
    int pidx = person_idx[rowg];
    const uint4* pp = (const uint4*)(peF16 + (size_t)pidx * 64 + h * 32);
    uint4* dp = (uint4*)(crow + 64 + h * 32);
    dp[0] = pp[0]; dp[1] = pp[1]; dp[2] = pp[2]; dp[3] = pp[3];

    float tv = tvec[rowg];
    const float* wt1 = (const float*)(smem + LDS_WT1);
    const float* b1t = (const float*)(smem + LDS_BT1);
    const int j0 = h * 16;
    #pragma unroll
    for (int jj = 0; jj < 16; jj += 4) {
      v4h hv = { (_Float16)fmaxf(fmaf(tv, wt1[j0 + jj + 0], b1t[j0 + jj + 0]), 0.0f),
                 (_Float16)fmaxf(fmaf(tv, wt1[j0 + jj + 1], b1t[j0 + jj + 1]), 0.0f),
                 (_Float16)fmaxf(fmaf(tv, wt1[j0 + jj + 2], b1t[j0 + jj + 2]), 0.0f),
                 (_Float16)fmaxf(fmaf(tv, wt1[j0 + jj + 3], b1t[j0 + jj + 3]), 0.0f) };
      *(v4h*)(crow + 128 + j0 + jj) = hv;
    }
  }
  __builtin_amdgcn_wave_barrier();

  // ---- load the 5 A-fragments of `combined` ONCE (reused by Gram + GEMM1) ----
  v16h aC[5];
  #pragma unroll
  for (int ks = 0; ks < 5; ++ks) aC[ks] = load_afrag(combW, 160, lane, ks * 32);

  // ---- Gram GEMM: G = state(16x64) . ze^T(64x8 pad16) -> gW ----
  {
    const _Float16* wf = (const _Float16*)(smem + LDS_ZET);
    v8f c;
    #pragma unroll
    for (int q = 0; q < 8; ++q) c[q] = 0.0f;
    c = __builtin_amdgcn_wmma_f32_16x16x32_f16(false, aC[0], false, load_bfrag(wf, 0, lane), (short)0, c, false, false);
    c = __builtin_amdgcn_wmma_f32_16x16x32_f16(false, aC[1], false, load_bfrag(wf, 1, lane), (short)0, c, false, false);
    #pragma unroll
    for (int q = 0; q < 8; ++q) gW[(m0 + q) * 16 + ncol] = c[q];
  }

  // ---- GEMM1: [16x160] x [160x128] + bf1', ReLU -> h1 (A reused from regs) ----
  {
    const _Float16* wf = (const _Float16*)(smem + LDS_WF1);
    const float* b1 = (const float*)(smem + LDS_BF1);
    for (int nt = 0; nt < 8; ++nt) {
      float bias = b1[nt * 16 + ncol];
      v8f c;
      #pragma unroll
      for (int q = 0; q < 8; ++q) c[q] = bias;
      #pragma unroll
      for (int ks = 0; ks < 5; ++ks) {
        v16h b = load_bfrag(wf, nt * 5 + ks, lane);
        c = __builtin_amdgcn_wmma_f32_16x16x32_f16(false, aC[ks], false, b, (short)0, c, false, false);
      }
      #pragma unroll
      for (int q = 0; q < 8; ++q)
        h1W[(m0 + q) * 128 + nt * 16 + ncol] = (_Float16)fmaxf(c[q], 0.0f);
    }
  }
  __builtin_amdgcn_wave_barrier();

  // ---- GEMM2: [16x128] x [128x64] + bf2, ReLU -> h2 (A hoisted to regs) ----
  {
    const _Float16* wf = (const _Float16*)(smem + LDS_WF2);
    const float* b2 = (const float*)(smem + LDS_BF2);
    v16h aH[4];
    #pragma unroll
    for (int ks = 0; ks < 4; ++ks) aH[ks] = load_afrag(h1W, 128, lane, ks * 32);
    for (int nt = 0; nt < 4; ++nt) {
      float bias = b2[nt * 16 + ncol];
      v8f c;
      #pragma unroll
      for (int q = 0; q < 8; ++q) c[q] = bias;
      #pragma unroll
      for (int ks = 0; ks < 4; ++ks) {
        v16h b = load_bfrag(wf, nt * 4 + ks, lane);
        c = __builtin_amdgcn_wmma_f32_16x16x32_f16(false, aH[ks], false, b, (short)0, c, false, false);
      }
      #pragma unroll
      for (int q = 0; q < 8; ++q)
        h2W[(m0 + q) * 64 + nt * 16 + ncol] = (_Float16)fmaxf(c[q], 0.0f);
    }
  }
  __builtin_amdgcn_wave_barrier();

  // ---- GEMM3: [16x64] x [64x8(pad16)] + bf3 -> desired_flows ----
  {
    const _Float16* wf = (const _Float16*)(smem + LDS_WF3);
    const float* b3 = (const float*)(smem + LDS_BF3);
    float bias = b3[ncol];
    v8f c;
    #pragma unroll
    for (int q = 0; q < 8; ++q) c[q] = bias;
    #pragma unroll
    for (int ks = 0; ks < 2; ++ks) {
      v16h a = load_afrag(h2W, 64, lane, ks * 32);
      v16h b = load_bfrag(wf, ks, lane);
      c = __builtin_amdgcn_wmma_f32_16x16x32_f16(false, a, false, b, (short)0, c, false, false);
    }
    #pragma unroll
    for (int q = 0; q < 8; ++q) desW[(m0 + q) * 16 + ncol] = c[q];
  }
  __builtin_amdgcn_wave_barrier();

  // ---- epilogue: dist^2 = |s|^2 - 2G + |ze|^2, softmax, flows, velocity ----
  {
    const float* zel  = (const float*)(smem + LDS_ZE);
    const float* adjl = (const float*)(smem + LDS_ADJ);
    const float* znl  = (const float*)(smem + LDS_ZN);
    float stl[32];
    const float4* sp = (const float4*)(state + (size_t)rowg * 64 + h * 32);
    #pragma unroll
    for (int q = 0; q < 8; ++q) {
      float4 v = sp[q];
      stl[4*q] = v.x; stl[4*q+1] = v.y; stl[4*q+2] = v.z; stl[4*q+3] = v.w;
    }
    float s2 = 0.0f;
    #pragma unroll
    for (int d2 = 0; d2 < 32; ++d2) s2 = fmaf(stl[d2], stl[d2], s2);
    s2 += __shfl_xor(s2, 16, 32);

    float lg[8]; float mx = -3.0e38f;
    #pragma unroll
    for (int z = 0; z < 8; ++z) {
      float dd = fmaxf(s2 + znl[z] - 2.0f * gW[r * 16 + z], 0.0f);
      float lz = -10.0f * sqrtf(dd);
      lg[z] = lz; mx = fmaxf(mx, lz);
    }
    float w[8]; float se = 0.0f;
    #pragma unroll
    for (int z = 0; z < 8; ++z) { w[z] = __expf(lg[z] - mx); se += w[z]; }
    float inv = 1.0f / se;
    #pragma unroll
    for (int z = 0; z < 8; ++z) { float wz = w[z] * inv; w[z] = (wz > 0.01f) ? wz : 0.0f; }
    float flows[8]; float fsum = 0.0f;
    #pragma unroll
    for (int z = 0; z < 8; ++z) {
      float a = 0.0f;
      #pragma unroll
      for (int z2 = 0; z2 < 8; ++z2) a = fmaf(w[z2], adjl[z2 * 8 + z], a);
      flows[z] = desW[r * 16 + z] * a;
      fsum += flows[z];
    }
    float vb[32];
    #pragma unroll
    for (int d2 = 0; d2 < 32; ++d2) {
      float acc = 0.0f;
      #pragma unroll
      for (int z = 0; z < 8; ++z) acc = fmaf(flows[z], zel[z * 64 + h * 32 + d2], acc);
      vb[d2] = 0.01f * (acc - fsum * stl[d2]);
    }
    float4* op = (float4*)(out + (size_t)rowg * 64 + h * 32);
    #pragma unroll
    for (int q = 0; q < 8; ++q) op[q] = make_float4(vb[4*q], vb[4*q+1], vb[4*q+2], vb[4*q+3]);
  }
}

extern "C" void kernel_launch(void* const* d_in, const int* in_sizes, int n_in,
                              void* d_out, int out_size, void* d_ws, size_t ws_size,
                              hipStream_t stream) {
  (void)in_sizes; (void)n_in; (void)out_size; (void)ws_size;
  const float* state      = (const float*)d_in[0];
  const float* t          = (const float*)d_in[1];
  const float* person_x   = (const float*)d_in[2];
  const int*   person_idx = (const int*)d_in[3];
  const float* zone_emb   = (const float*)d_in[4];
  const float* adjacency  = (const float*)d_in[5];
  const float* Wp1 = (const float*)d_in[6];
  const float* bp1 = (const float*)d_in[7];
  const float* Wp2 = (const float*)d_in[8];
  const float* bp2 = (const float*)d_in[9];
  const float* Wt1 = (const float*)d_in[10];
  const float* bt1 = (const float*)d_in[11];
  const float* Wt2 = (const float*)d_in[12];
  const float* bt2 = (const float*)d_in[13];
  const float* Wf1 = (const float*)d_in[14];
  const float* bf1 = (const float*)d_in[15];
  const float* Wf2 = (const float*)d_in[16];
  const float* bf2 = (const float*)d_in[17];
  const float* Wf3 = (const float*)d_in[18];
  const float* bf3 = (const float*)d_in[19];
  float* out = (float*)d_out;

  _Float16* peF16 = (_Float16*)d_ws;                          // 640000 halfs
  _Float16* frags = (_Float16*)((char*)d_ws + 1280000);       // 30720 halfs
  float*    W1cat = (float*)((char*)d_ws + 1341440);          // 160x128 f32
  float*    bf1p  = W1cat + 20480;                            // 128 f32

  person_embed_kernel<<<NPERS, 64, 0, stream>>>(person_x, Wp1, bp1, Wp2, bp2, peF16);
  fuse_time_kernel<<<81, 256, 0, stream>>>(Wf1, bf1, Wt2, bt2, W1cat, bf1p);
  weight_swizzle_kernel<<<120, 256, 0, stream>>>(W1cat, Wf2, Wf3, zone_emb, frags);
  ode_main_kernel<<<B_TOTAL / 128, 256, LDS_TOTAL, stream>>>(
      state, t, person_idx, zone_emb, adjacency,
      Wt1, bt1, bf1p, bf2, bf3, peF16, frags, out);
}
